// T_Moduel_57921928953935
// MI455X (gfx1250) — matrix operand order repaired and checked by reading, compiled, not run on hardware
//
#include <hip/hip_runtime.h>

// ---------------------------------------------------------------------------
// Fused criss-cross-style spatio-temporal attention for MI455X (gfx1250).
//
// Roofline: ~11 GFLOP vs ~2.3 GB HBM traffic @23.3 TB/s => memory bound.
// Single fused kernel: read x once per tile into LDS (f16), all
// intermediates on-chip, write out once. Projections and the value GEMM use
// v_wmma_f32_16x16x32_f16. The value projection is commuted with the
// attention sum (out = (sum attn * x_shift) @ Wp^T) so p is never
// materialized. v3: attention aggregation uses packed f16 FMA
// (v_pk_fma_f16) -- y feeds an f16 WMMA anyway, and this halves the
// dominant VALU stream (45x64 scalar FMA -> 45x32 packed FMA per pixel).
// ---------------------------------------------------------------------------

typedef _Float16 v16h __attribute__((ext_vector_type(16)));
typedef _Float16 v8h  __attribute__((ext_vector_type(8)));
typedef _Float16 h2   __attribute__((ext_vector_type(2)));
typedef float    v8f  __attribute__((ext_vector_type(8)));

#define BB     8
#define TT     5
#define CC     64
#define HH     128
#define WW     128
#define HWSZ   (HH*WW)

#define TILE_H 8
#define TILE_W 16
#define HALO_W (TILE_W+2)          // 18
#define HALO_H (TILE_H+2)          // 10
#define NHALO  (HALO_H*HALO_W)     // 180 real halo pixels
#define NSLOT  192                 // padded to multiple of 16 for WMMA groups
#define NPIX   (TILE_H*TILE_W)     // 128 interior pixels
#define STG_PITCH 129              // f32 staging pitch (bank-friendly)

// dynamic LDS layout (bytes)
#define X_OFF   0
#define X_SZ    (TT*NSLOT*CC*2)            // 122880: x tile f16 [t][slot][c]
#define QK_OFF  (X_OFF + X_SZ)
#define QK_SZ   (TT*NSLOT*16*4)            // 61440: qk f32 [t][slot][16]
#define BP_OFF  (QK_OFF + QK_SZ)
#define BP_SZ   (8*32*16*2)                // 8192: Wp B-operands [op][lane][16] f16
#define Y_OFF   (BP_OFF + BP_SZ)
#define Y_SZ    (CC*STG_PITCH*4)           // 33024: y f16 [pix][c] / f32 staging
#define LDS_TOTAL (Y_OFF + Y_SZ)           // 225536 B < 320KB WGP LDS

__device__ __forceinline__ v8f wmma16(v16h a, v16h b, v8f c) {
  // (neg_a, A, neg_b, B, c_mod, C, reuse_a, reuse_b)
  return __builtin_amdgcn_wmma_f32_16x16x32_f16(false, a, false, b, (short)0, c,
                                                false, false);
}

// 16-lane-striped A operand (16 pixels x 32 K f16) from a 64-ch LDS row.
// Contiguous K order (matches B): lanes 0-15 hold K=ko..ko+15,
// lanes 16-31 hold K=ko+16..ko+31.
__device__ __forceinline__ v16h load_a(const _Float16* row, int ko, int hi) {
  return *(const v16h*)(row + ko + hi * 16);
}

__global__ __launch_bounds__(256, 1)
void cca_fused_kernel(const float* __restrict__ x, const float* __restrict__ Wq,
                      const float* __restrict__ Wk, const float* __restrict__ Wp,
                      const float* __restrict__ gamma, float* __restrict__ out) {
  extern __shared__ __align__(32) char lds[];
  _Float16* xl  = (_Float16*)(lds + X_OFF);   // [t][slot][64] f16
  float*    qk  = (float*)(lds + QK_OFF);     // [t][slot][16] f32 (q:0-7, k:8-15)
  _Float16* bpl = (_Float16*)(lds + BP_OFF);  // [8 ops][32 lanes][16] f16
  _Float16* yh  = (_Float16*)(lds + Y_OFF);   // [pix][64] f16
  float*    stg = (float*)(lds + Y_OFF);      // [c][STG_PITCH] f32 (reuses y)

  const int tid  = threadIdx.x;
  const int lane = tid & 31;     // wave32
  const int wave = tid >> 5;     // 8 waves
  const int b    = blockIdx.z;
  const int h0   = blockIdx.y * TILE_H;
  const int w0   = blockIdx.x * TILE_W;

  const int n  = lane & 15;      // WMMA N / C-matrix column for this lane
  const int hi = lane >> 4;      // lane-half selector for A/B K chunks

  // ------------------------------------------------------------------
  // bqk B-operand in registers (dead after projection phase).
  // B order used here (contiguous K, same as load_a): lanes 0-15 -> K=0..15, lanes 16-31 -> K=16..31.
  // ------------------------------------------------------------------
  v16h bqk[2];
  {
    const float* wrow = (n < 8) ? (Wq + n * CC) : (Wk + (n - 8) * CC);
#pragma unroll
    for (int ko = 0; ko < 2; ++ko) {
      v16h tv;
#pragma unroll
      for (int e = 0; e < 16; ++e) tv[e] = (_Float16)wrow[ko * 32 + hi * 16 + e];
      bqk[ko] = tv;
    }
  }

  // Wp B-operands are wave-invariant (lane-dependent only): build once
  // into LDS with wave 0, stream them back with ds_load_b128 at use sites.
  if (wave == 0) {
#pragma unroll
    for (int nt = 0; nt < 4; ++nt) {
      const float* prow = Wp + (nt * 16 + n) * CC;   // B[k][n] = Wp[n][k]
#pragma unroll
      for (int ko = 0; ko < 2; ++ko) {
        v16h tv;
#pragma unroll
        for (int e = 0; e < 16; ++e) tv[e] = (_Float16)prow[ko * 32 + hi * 16 + e];
        *(v16h*)(bpl + ((nt * 2 + ko) * 32 + lane) * 16) = tv;
      }
    }
  }

  // ------------------------------------------------------------------
  // Stage x tile (all 5 frames, zero-padded halo) into LDS as f16.
  // Consecutive threads -> consecutive w: coalesced HBM reads.
  // ------------------------------------------------------------------
  for (int idx = tid; idx < TT * CC * NHALO; idx += 256) {
    int t   = idx / (CC * NHALO);
    int r   = idx - t * (CC * NHALO);
    int c   = r / NHALO;
    int pix = r - c * NHALO;
    int hp  = pix / HALO_W, wp = pix - hp * HALO_W;
    int gh  = h0 - 1 + hp, gw = w0 - 1 + wp;
    float v = 0.f;
    if (gh >= 0 && gh < HH && gw >= 0 && gw < WW)
      v = x[(((size_t)b * TT + t) * CC + c) * HWSZ + (size_t)gh * WW + gw];
    xl[(t * NSLOT + pix) * CC + c] = (_Float16)v;
  }
  // zero the padded slots [NHALO, NSLOT) so garbage never feeds WMMA
  for (int idx = tid; idx < TT * (NSLOT - NHALO) * CC; idx += 256) {
    int t  = idx / ((NSLOT - NHALO) * CC);
    int r  = idx - t * ((NSLOT - NHALO) * CC);
    int sl = NHALO + r / CC;
    int c  = r - (r / CC) * CC;
    xl[(t * NSLOT + sl) * CC + c] = (_Float16)0.f;
  }
  __syncthreads();

  // ------------------------------------------------------------------
  // Projection GEMM via WMMA: qk[slot][0:16] = x[slot][0:64] @ [Wq^T|Wk^T]
  // 60 groups of 16 pixels across 8 waves, 2 K-steps each.
  // ------------------------------------------------------------------
  for (int g = wave; g < TT * (NSLOT / 16); g += 8) {
    int t    = g / (NSLOT / 16);
    int base = (g - t * (NSLOT / 16)) * 16;
    const _Float16* arow = xl + (t * NSLOT + base + n) * CC;  // pixel row M
    v8f acc;
#pragma unroll
    for (int r = 0; r < 8; ++r) acc[r] = 0.f;
#pragma unroll
    for (int ko = 0; ko < 2; ++ko)
      acc = wmma16(load_a(arow, ko * 32, hi), bqk[ko], acc);
    // C layout: lanes0-15 -> M=vgpr, lanes16-31 -> M=vgpr+8, N=lane%16
#pragma unroll
    for (int r = 0; r < 8; ++r)
      qk[(t * NSLOT + base + r + hi * 8) * 16 + n] = acc[r];
  }
  __syncthreads();

  const float gm = gamma[0];

  // ------------------------------------------------------------------
  // Per query frame: energies -> softmax(45) -> y = sum attn * x_shift
  // (packed f16 FMA), then out = gamma * (y @ Wp^T) + x  (WMMA GEMM).
  // ------------------------------------------------------------------
  for (int t = 0; t < TT; ++t) {
    __syncthreads();  // y/staging region reuse across t iterations

    if (tid < NPIX) {
      const int pix = tid;
      const int ih = pix >> 4, iw = pix & 15;
      const int qslot = (ih + 1) * HALO_W + (iw + 1);

      float q[8];
#pragma unroll
      for (int d = 0; d < 8; ++d) q[d] = qk[(t * NSLOT + qslot) * 16 + d];

      float e[TT][9];
      float emax = -1e30f;
#pragma unroll
      for (int s = 0; s < TT; ++s) {
#pragma unroll
        for (int k9 = 0; k9 < 9; ++k9) {
          const int dh = k9 / 3 - 1, dw = k9 % 3 - 1;
          const int kslot = (ih + 1 + dh) * HALO_W + (iw + 1 + dw);
          const float* kp = qk + (s * NSLOT + kslot) * 16 + 8;
          float acc = 0.f;
#pragma unroll
          for (int d = 0; d < 8; ++d) acc += q[d] * kp[d];
          e[s][k9] = acc;
          emax = fmaxf(emax, acc);
        }
      }
      float esum = 0.f;
#pragma unroll
      for (int s = 0; s < TT; ++s)
#pragma unroll
        for (int k9 = 0; k9 < 9; ++k9) {
          e[s][k9] = __expf(e[s][k9] - emax);
          esum += e[s][k9];
        }
      const float inv = 1.f / esum;

      // y accumulated as 32 packed f16 pairs -> v_pk_fma_f16 stream.
      h2 y2[CC / 2];
#pragma unroll
      for (int c2 = 0; c2 < CC / 2; ++c2) {
        y2[c2][0] = (_Float16)0.f;
        y2[c2][1] = (_Float16)0.f;
      }
#pragma unroll
      for (int s = 0; s < TT; ++s) {
#pragma unroll
        for (int k9 = 0; k9 < 9; ++k9) {
          const _Float16 wh = (_Float16)(e[s][k9] * inv);
          h2 w2; w2[0] = wh; w2[1] = wh;
          const int dh = k9 / 3 - 1, dw = k9 % 3 - 1;
          const int kslot = (ih + 1 + dh) * HALO_W + (iw + 1 + dw);
          const _Float16* xp = xl + (s * NSLOT + kslot) * CC;
#pragma unroll
          for (int c8 = 0; c8 < 8; ++c8) {       // 16B ds loads
            v8h xv = *(const v8h*)(xp + c8 * 8);
#pragma unroll
            for (int j = 0; j < 4; ++j) {
              h2 xp2 = __builtin_shufflevector(xv, xv, -1, -1);
              xp2[0] = xv[2 * j];
              xp2[1] = xv[2 * j + 1];
              y2[c8 * 4 + j] += w2 * xp2;        // v_pk_fma_f16
            }
          }
        }
      }
      // pack 4 h2 -> 16B and store y row (ds_store_b128)
#pragma unroll
      for (int g8 = 0; g8 < 8; ++g8) {
        v8h o;
#pragma unroll
        for (int j = 0; j < 4; ++j) {
          o[2 * j]     = y2[g8 * 4 + j][0];
          o[2 * j + 1] = y2[g8 * 4 + j][1];
        }
        *(v8h*)(yh + pix * CC + g8 * 8) = o;
      }
    }
    __syncthreads();

    // out tile = y @ Wp^T : wave g handles pixel group g (16 pixels), 4 N-tiles
    v8f acc2[4];
    {
      const int base = wave * 16;
      const _Float16* arow = yh + (base + n) * CC;
#pragma unroll
      for (int nt = 0; nt < 4; ++nt)
#pragma unroll
        for (int r = 0; r < 8; ++r) acc2[nt][r] = 0.f;
#pragma unroll
      for (int ko = 0; ko < 2; ++ko) {
        v16h a = load_a(arow, ko * 32, hi);
#pragma unroll
        for (int nt = 0; nt < 4; ++nt) {
          v16h bpv = *(const v16h*)(bpl + ((nt * 2 + ko) * 32 + lane) * 16);
          acc2[nt] = wmma16(a, bpv, acc2[nt]);
        }
      }
    }
    __syncthreads();  // everyone done reading yh before f32 staging overwrite
    {
      const int base = wave * 16;
#pragma unroll
      for (int nt = 0; nt < 4; ++nt)
#pragma unroll
        for (int r = 0; r < 8; ++r)
          stg[(nt * 16 + n) * STG_PITCH + base + r + hi * 8] = acc2[nt][r];
    }
    __syncthreads();

    // gamma * out + x, coalesced (consecutive lanes -> consecutive w)
    for (int idx = tid; idx < CC * NPIX; idx += 256) {
      const int c   = idx >> 7;
      const int pix = idx & 127;
      const int ih = pix >> 4, iw = pix & 15;
      const size_t ga = (((size_t)b * TT + t) * CC + c) * HWSZ +
                        (size_t)(h0 + ih) * WW + (w0 + iw);
      out[ga] = gm * stg[c * STG_PITCH + pix] + x[ga];
    }
  }
}

extern "C" void kernel_launch(void* const* d_in, const int* in_sizes, int n_in,
                              void* d_out, int out_size, void* d_ws, size_t ws_size,
                              hipStream_t stream) {
  (void)in_sizes; (void)n_in; (void)out_size; (void)d_ws; (void)ws_size;
  const float* x     = (const float*)d_in[0];
  const float* Wq    = (const float*)d_in[1];
  const float* Wk    = (const float*)d_in[2];
  const float* Wp    = (const float*)d_in[3];
  const float* gamma = (const float*)d_in[4];
  float* out = (float*)d_out;

  // 225.5KB dynamic LDS (< 320KB WGP limit) needs the opt-in attribute.
  hipFuncSetAttribute((const void*)cca_fused_kernel,
                      hipFuncAttributeMaxDynamicSharedMemorySize, LDS_TOTAL);

  dim3 grid(WW / TILE_W, HH / TILE_H, BB);  // 8 x 16 x 8 = 1024 workgroups
  cca_fused_kernel<<<grid, 256, LDS_TOTAL, stream>>>(x, Wq, Wk, Wp, gamma, out);
}
